// MambaBlock_41936060678336
// MI455X (gfx1250) — compile-verified, matrix operand order checked by reading, not run on hardware
//
#include <hip/hip_runtime.h>
#include <hip/hip_bf16.h>
#include <math.h>

// ---------------------------------------------------------------------------
// Mamba block for MI455X (gfx1250, wave32, WMMA).
//   D_MODEL = D_INNER = 768, D_STATE = 64, L = 1024, B = 1, K(conv) = 4
// ---------------------------------------------------------------------------

#define D_MODEL 768
#define D_INNER 768
#define D_STATE 64
#define SEQ     1024
#define KCONV   4
#define NBIG    (D_INNER * D_STATE)   // 49152
#define KDIM    768                   // reduction dim of every GEMM here
#define LN_EPS  1e-5f

typedef __bf16 bf16;
typedef __attribute__((ext_vector_type(16))) __bf16 v16bf;
typedef __attribute__((ext_vector_type(8)))  float  v8f;

union FragU {
    uint4  u[2];
    v16bf  v;
    bf16   e[16];
};

// ---------------------------------------------------------------------------
// Kernel 1: LayerNorm over d_model per timestep.
// ---------------------------------------------------------------------------
__global__ void ln_kernel(const float* __restrict__ x,
                          const float* __restrict__ g,
                          const float* __restrict__ b,
                          float* __restrict__ xn)
{
    const int t   = blockIdx.x;
    const int tid = threadIdx.x;
    __shared__ float red[256];

    float s = 0.f;
    for (int d = tid; d < D_MODEL; d += 256) s += x[t * D_MODEL + d];
    red[tid] = s; __syncthreads();
    for (int o = 128; o > 0; o >>= 1) {
        if (tid < o) red[tid] += red[tid + o];
        __syncthreads();
    }
    const float mu = red[0] * (1.0f / D_MODEL);
    __syncthreads();

    float s2 = 0.f;
    for (int d = tid; d < D_MODEL; d += 256) {
        float v = x[t * D_MODEL + d] - mu;
        s2 += v * v;
    }
    red[tid] = s2; __syncthreads();
    for (int o = 128; o > 0; o >>= 1) {
        if (tid < o) red[tid] += red[tid + o];
        __syncthreads();
    }
    const float rstd = rsqrtf(red[0] * (1.0f / D_MODEL) + LN_EPS);

    for (int d = tid; d < D_MODEL; d += 256) {
        xn[t * D_MODEL + d] = (x[t * D_MODEL + d] - mu) * rstd * g[d] + b[d];
    }
}

// ---------------------------------------------------------------------------
// Kernel 2: depthwise causal conv1d (K=4, left pad 3) -> bf16 activations.
// ---------------------------------------------------------------------------
__global__ void conv_kernel(const float* __restrict__ xn,
                            const float* __restrict__ cw,
                            const float* __restrict__ cb,
                            bf16* __restrict__ xc_bf)
{
    const int idx = blockIdx.x * blockDim.x + threadIdx.x;
    if (idx >= SEQ * D_INNER) return;
    const int t = idx / D_INNER;
    const int d = idx - t * D_INNER;

    float acc = cb[d];
    #pragma unroll
    for (int k = 0; k < KCONV; ++k) {
        const int ts = t - (KCONV - 1) + k;
        if (ts >= 0) acc += xn[ts * D_INNER + d] * cw[d * KCONV + k];
    }
    xc_bf[idx] = (bf16)acc;
}

// ---------------------------------------------------------------------------
// fp32x16 -> v16bf conversion (v_cvt_pk_bf16_f32)
// ---------------------------------------------------------------------------
__device__ __forceinline__ v16bf cvt_frag(const float4& f0, const float4& f1,
                                          const float4& f2, const float4& f3)
{
    FragU r;
    r.e[ 0] = (bf16)f0.x; r.e[ 1] = (bf16)f0.y; r.e[ 2] = (bf16)f0.z; r.e[ 3] = (bf16)f0.w;
    r.e[ 4] = (bf16)f1.x; r.e[ 5] = (bf16)f1.y; r.e[ 6] = (bf16)f1.z; r.e[ 7] = (bf16)f1.w;
    r.e[ 8] = (bf16)f2.x; r.e[ 9] = (bf16)f2.y; r.e[10] = (bf16)f2.z; r.e[11] = (bf16)f2.w;
    r.e[12] = (bf16)f3.x; r.e[13] = (bf16)f3.y; r.e[14] = (bf16)f3.z; r.e[15] = (bf16)f3.w;
    return r.v;
}

// ---------------------------------------------------------------------------
// WMMA GEMM: out[M,N] = A[M,K](bf16) * W[N,K]^T (fp32 weights, cvt on load).
//
// One wave owns a 32(M) x 64(N) strip = 2 M-tiles x 4 N-tiles.
// K is compile-time (768): all W/A fragments use immediate offsets off two
// base pointers -> one load clause per K step, no address VALU chains.
// __launch_bounds__(256, 2): 2 waves/SIMD VGPR budget -> no scratch spills
// for the ~190 live VGPRs (64 accum + 80 raw + fragments).
// Strips are N-major so consecutive waves/blocks share the same 64 W columns
// while sweeping M (W streamed from HBM ~once, re-served from L2).
// EPI: 0 = fp32 store, 1 = softplus(v + bias) fp32, 2 = bf16 store.
// ---------------------------------------------------------------------------
template <int EPI>
__global__ __launch_bounds__(256, 2)
void wmma_gemm_kernel(const bf16*  __restrict__ A,
                      const float* __restrict__ W,
                      float*       __restrict__ outF,
                      bf16*        __restrict__ outB,
                      const float* __restrict__ bias,
                      int M, int N)
{
    const int lane  = threadIdx.x & 31;
    const int wave  = threadIdx.x >> 5;
    const int strip = blockIdx.x * 8 + wave;
    const int mStrips = M >> 5;                 // 32 rows per wave
    const int nStrips = N >> 6;                 // 64 cols per wave
    if (strip >= mStrips * nStrips) return;

    const int nBase = (strip / mStrips) << 6;   // N-major ordering
    const int mBase = (strip % mStrips) << 5;

    const int kb = (lane >> 4) * 8;             // K-half select (ISA 7.12.2)
    const int lr = lane & 15;

    // single base pointer each for A and W; fragment offsets are immediates
    const bf16*  pA = A + (size_t)(mBase + lr) * KDIM + kb;
    const float* pW = W + (size_t)(nBase + lr) * KDIM + kb;

    const v8f vzero = {0.f,0.f,0.f,0.f,0.f,0.f,0.f,0.f};
    v8f acc00 = vzero, acc01 = vzero, acc02 = vzero, acc03 = vzero;
    v8f acc10 = vzero, acc11 = vzero, acc12 = vzero, acc13 = vzero;

    #define AOFF (16 * KDIM)                    // next M-tile row block
    #define WOFF (16 * KDIM)                    // next N-tile col block

    for (int kk = 0; kk < KDIM; kk += 32) {
        // ---- batched raw loads: 20 independent b128 loads, imm offsets ----
        const uint4  a00 = *(const uint4*)(pA);
        const uint4  a01 = *(const uint4*)(pA + 16);
        const uint4  a10 = *(const uint4*)(pA + AOFF);
        const uint4  a11 = *(const uint4*)(pA + AOFF + 16);

        const float4 w00 = *(const float4*)(pW);
        const float4 w01 = *(const float4*)(pW + 4);
        const float4 w02 = *(const float4*)(pW + 16);
        const float4 w03 = *(const float4*)(pW + 20);
        const float4 w10 = *(const float4*)(pW + WOFF);
        const float4 w11 = *(const float4*)(pW + WOFF + 4);
        const float4 w12 = *(const float4*)(pW + WOFF + 16);
        const float4 w13 = *(const float4*)(pW + WOFF + 20);
        const float4 w20 = *(const float4*)(pW + 2 * WOFF);
        const float4 w21 = *(const float4*)(pW + 2 * WOFF + 4);
        const float4 w22 = *(const float4*)(pW + 2 * WOFF + 16);
        const float4 w23 = *(const float4*)(pW + 2 * WOFF + 20);
        const float4 w30 = *(const float4*)(pW + 3 * WOFF);
        const float4 w31 = *(const float4*)(pW + 3 * WOFF + 4);
        const float4 w32 = *(const float4*)(pW + 3 * WOFF + 16);
        const float4 w33 = *(const float4*)(pW + 3 * WOFF + 20);

        // prefetch next K-slab of the streamed weights (global_prefetch_b8)
        if (kk + 32 < KDIM) {
            __builtin_prefetch((const void*)(pW + 32), 0, 1);
            __builtin_prefetch((const void*)(pW + 2 * WOFF + 32), 0, 1);
        }

        pA += 32;
        pW += 32;

        // ---- assemble fragments ----
        FragU fa0; fa0.u[0] = a00; fa0.u[1] = a01;
        FragU fa1; fa1.u[0] = a10; fa1.u[1] = a11;
        const v16bf af0 = fa0.v;
        const v16bf af1 = fa1.v;

        const v16bf bf0 = cvt_frag(w00, w01, w02, w03);
        const v16bf bf1 = cvt_frag(w10, w11, w12, w13);
        const v16bf bf2 = cvt_frag(w20, w21, w22, w23);
        const v16bf bf3 = cvt_frag(w30, w31, w32, w33);

        // ---- 8 WMMAs; each B fragment reused for 2 M-tiles ----
        acc00 = __builtin_amdgcn_wmma_f32_16x16x32_bf16(false, af0, false, bf0, (short)0, acc00, false, false);
        acc10 = __builtin_amdgcn_wmma_f32_16x16x32_bf16(false, af1, false, bf0, (short)0, acc10, false, false);
        acc01 = __builtin_amdgcn_wmma_f32_16x16x32_bf16(false, af0, false, bf1, (short)0, acc01, false, false);
        acc11 = __builtin_amdgcn_wmma_f32_16x16x32_bf16(false, af1, false, bf1, (short)0, acc11, false, false);
        acc02 = __builtin_amdgcn_wmma_f32_16x16x32_bf16(false, af0, false, bf2, (short)0, acc02, false, false);
        acc12 = __builtin_amdgcn_wmma_f32_16x16x32_bf16(false, af1, false, bf2, (short)0, acc12, false, false);
        acc03 = __builtin_amdgcn_wmma_f32_16x16x32_bf16(false, af0, false, bf3, (short)0, acc03, false, false);
        acc13 = __builtin_amdgcn_wmma_f32_16x16x32_bf16(false, af1, false, bf3, (short)0, acc13, false, false);
    }
    #undef AOFF
    #undef WOFF

    // ---- epilogue: C/D layout VGPR r -> M = r + 8*(lane>=16), N = lane&15 --
    const int mOff = (lane >> 4) << 3;
    auto store_tile = [&](const v8f& acc, int mT, int nT) {
        #pragma unroll
        for (int r = 0; r < 8; ++r) {
            const int row = mBase + mT * 16 + mOff + r;
            const int col = nBase + nT * 16 + lr;
            float v = acc[r];
            if (EPI == 0) {
                outF[(size_t)row * N + col] = v;
            } else if (EPI == 1) {
                v += bias[col];
                outF[(size_t)row * N + col] = (v > 20.f) ? v : log1pf(__expf(v));
            } else {
                outB[(size_t)row * N + col] = (bf16)v;
            }
        }
    };
    store_tile(acc00, 0, 0); store_tile(acc01, 0, 1);
    store_tile(acc02, 0, 2); store_tile(acc03, 0, 3);
    store_tile(acc10, 1, 0); store_tile(acc11, 1, 1);
    store_tile(acc12, 1, 2); store_tile(acc13, 1, 3);
}

// ---------------------------------------------------------------------------
// Selective scan, fused: Ad = exp(delta*A) and Bx = Bm*x_inner computed
// in-register (never materialized).  One wave per channel d; each lane owns
// states n and n+32; y = sum_n C*h via __shfl_xor; SiLU applied, bf16 out.
// ---------------------------------------------------------------------------
__global__ void scan_kernel(const float* __restrict__ delta,
                            const float* __restrict__ x_inner,
                            const bf16*  __restrict__ Bm,
                            const bf16*  __restrict__ Cm,
                            const float* __restrict__ A_log,
                            bf16* __restrict__ ysilu)
{
    const int d    = blockIdx.x;        // 0..767
    const int lane = threadIdx.x;       // 0..31
    const int n0 = lane, n1 = lane + 32;

    const float A0 = -__expf(A_log[d * D_STATE + n0]);
    const float A1 = -__expf(A_log[d * D_STATE + n1]);

    float h0 = 0.f, h1 = 0.f;
    const size_t chanOff = (size_t)d * D_STATE;

    for (int t = 0; t < SEQ; ++t) {
        const float dt = delta[t * D_INNER + d];
        const float xi = x_inner[t * D_INNER + d];
        const size_t base = (size_t)t * NBIG + chanOff;

        const float b0 = (float)Bm[base + n0];
        const float b1 = (float)Bm[base + n1];
        const float c0 = (float)Cm[base + n0];
        const float c1 = (float)Cm[base + n1];

        h0 = __expf(dt * A0) * h0 + b0 * xi;
        h1 = __expf(dt * A1) * h1 + b1 * xi;

        float y = c0 * h0 + c1 * h1;
        #pragma unroll
        for (int off = 16; off >= 1; off >>= 1)
            y += __shfl_xor(y, off, 32);

        if (lane == 0) {
            const float s = y / (1.f + __expf(-y));     // silu
            ysilu[t * D_INNER + d] = (bf16)s;
        }
    }
}

// ---------------------------------------------------------------------------
// Host-side launch
// ---------------------------------------------------------------------------
static inline size_t align_up(size_t v) { return (v + 255) & ~(size_t)255; }

extern "C" void kernel_launch(void* const* d_in, const int* in_sizes, int n_in,
                              void* d_out, int out_size, void* d_ws, size_t ws_size,
                              hipStream_t stream)
{
    const float* x        = (const float*)d_in[0];
    const float* ln_g     = (const float*)d_in[1];
    const float* ln_b     = (const float*)d_in[2];
    const float* conv_w   = (const float*)d_in[3];
    const float* conv_b   = (const float*)d_in[4];
    const float* x_proj_w = (const float*)d_in[5];
    const float* dt_w     = (const float*)d_in[6];
    const float* dt_b     = (const float*)d_in[7];
    const float* A_log    = (const float*)d_in[8];
    const float* B_w      = (const float*)d_in[9];
    const float* C_w      = (const float*)d_in[10];
    const float* out_w    = (const float*)d_in[11];
    float* out = (float*)d_out;

    // workspace carve-up
    char* ws = (char*)d_ws;
    size_t off = 0;
    float* xn      = (float*)(ws + off); off = align_up(off + (size_t)SEQ * D_MODEL * 4);
    bf16*  xc_bf   = (bf16*) (ws + off); off = align_up(off + (size_t)SEQ * D_INNER * 2);
    float* x_inner = (float*)(ws + off); off = align_up(off + (size_t)SEQ * D_INNER * 4);
    float* delta   = (float*)(ws + off); off = align_up(off + (size_t)SEQ * D_INNER * 4);
    bf16*  Bm      = (bf16*) (ws + off); off = align_up(off + (size_t)SEQ * NBIG * 2);
    bf16*  Cm      = (bf16*) (ws + off); off = align_up(off + (size_t)SEQ * NBIG * 2);
    bf16*  ysilu   = (bf16*) (ws + off); off = align_up(off + (size_t)SEQ * D_INNER * 2);
    (void)ws_size; (void)n_in; (void)in_sizes; (void)out_size;

    // 1) LayerNorm
    ln_kernel<<<SEQ, 256, 0, stream>>>(x, ln_g, ln_b, xn);

    // 2) depthwise causal conv -> bf16 activations
    conv_kernel<<<(SEQ * D_INNER + 255) / 256, 256, 0, stream>>>(xn, conv_w, conv_b, xc_bf);

    // GEMM grid helper: one wave per 32x64 strip, 8 waves per block
    auto gemm_blocks = [](int M, int N) { return ((M >> 5) * (N >> 6) + 7) / 8; };

    // 3) x_inner = xc @ x_proj_w^T          (1024x768x768)
    wmma_gemm_kernel<0><<<gemm_blocks(SEQ, D_INNER), 256, 0, stream>>>(
        xc_bf, x_proj_w, x_inner, nullptr, nullptr, SEQ, D_INNER);

    // 4) delta = softplus(xc @ dt_w^T + dt_b)
    wmma_gemm_kernel<1><<<gemm_blocks(SEQ, D_INNER), 256, 0, stream>>>(
        xc_bf, dt_w, delta, nullptr, dt_b, SEQ, D_INNER);

    // 5) Bm = xc @ B_w^T, Cm = xc @ C_w^T   (1024x49152x768, bf16 out)
    wmma_gemm_kernel<2><<<gemm_blocks(SEQ, NBIG), 256, 0, stream>>>(
        xc_bf, B_w, nullptr, Bm, nullptr, SEQ, NBIG);
    wmma_gemm_kernel<2><<<gemm_blocks(SEQ, NBIG), 256, 0, stream>>>(
        xc_bf, C_w, nullptr, Cm, nullptr, SEQ, NBIG);

    // 6) fused selective scan + SiLU
    scan_kernel<<<D_INNER, 32, 0, stream>>>(delta, x_inner, Bm, Cm, A_log, ysilu);

    // 7) out = silu(y) @ out_w^T            (1024x768x768) -> d_out (fp32)
    wmma_gemm_kernel<0><<<gemm_blocks(SEQ, D_MODEL), 256, 0, stream>>>(
        ysilu, out_w, out, nullptr, nullptr, SEQ, D_MODEL);
}
